// TransformerBlock_3650722202172
// MI455X (gfx1250) — matrix operand order repaired and checked by reading, not compiled, not run on hardware
//
#include <hip/hip_runtime.h>
#include <hip/hip_bf16.h>

typedef unsigned short u16;
typedef __attribute__((ext_vector_type(16))) __bf16 v16bf;
typedef __attribute__((ext_vector_type(8)))  float  v8f;
typedef int async_b128_t __attribute__((vector_size(16)));   // int __vector(4)

// ----------------------------------------------------------- feature probes

#if defined(__has_builtin)
#  if __has_builtin(__builtin_amdgcn_global_load_async_to_lds_b128)
#    define GEMM_ASYNC 1
#  endif
#endif
#ifndef GEMM_ASYNC
#  define GEMM_ASYNC 0
#endif

// ---------------------------------------------------------------- utilities

static __device__ __forceinline__ u16 f32_to_bf16(float f) {
    unsigned int u = __float_as_uint(f);
    u = (u + 0x7FFFu + ((u >> 16) & 1u)) >> 16;   // round-to-nearest-even
    return (u16)u;
}

static __device__ __forceinline__ v8f v8f_zero() {
    v8f z;
#pragma unroll
    for (int i = 0; i < 8; ++i) z[i] = 0.f;
    return z;
}

static __device__ __forceinline__ v8f wmma_bf16(v16bf a, v16bf b, v8f c) {
    return __builtin_amdgcn_wmma_f32_16x16x32_bf16(
        false, a, false, b, (short)0, c, false, false);
}

// 16B global->LDS copy: async DMA (ASYNCcnt) when available, else sync.
static __device__ __forceinline__ void copy16(u16* dst_lds, const u16* src_glb) {
#if GEMM_ASYNC
    u16* s = const_cast<u16*>(src_glb);
    __builtin_amdgcn_global_load_async_to_lds_b128(
        (__attribute__((address_space(1))) async_b128_t*)s,
        (__attribute__((address_space(3))) async_b128_t*)dst_lds,
        0 /*imm offset*/, 0 /*cpol*/);
#else
    *reinterpret_cast<uint4*>(dst_lds) = *reinterpret_cast<const uint4*>(src_glb);
#endif
}

static __device__ __forceinline__ void stage_fence() {
#if GEMM_ASYNC
#  if defined(__has_builtin) && __has_builtin(__builtin_amdgcn_s_wait_asynccnt)
    __builtin_amdgcn_s_wait_asynccnt(0);
#  else
    asm volatile("s_wait_asynccnt 0" ::: "memory");
#  endif
#endif
    __syncthreads();
}

// A fragment: 16x32 bf16, lane(0-15)=row M; half 0: K{0..7,16..23}, half 1: +8.
static __device__ __forceinline__ v16bf load_a_frag(const u16* lds, int rowBase,
                                                    int stride, int k0, int lane) {
    int m = lane & 15, half = lane >> 4;
    const u16* p = lds + (rowBase + m) * stride + k0 + half * 8;
    union { uint4 q[2]; v16bf v; } f;
    f.q[0] = *reinterpret_cast<const uint4*>(p);
    f.q[1] = *reinterpret_cast<const uint4*>(p + 16);
    return f.v;
}

// B fragment: 32x16 bf16, lane(0-15)=col N; half 0: K{0..7,16..23}, half 1: +8.
// LDS layout is [N][K] row-major (bf16 units).
static __device__ __forceinline__ v16bf load_bt_frag(const u16* lds, int colBase,
                                                     int stride, int k0, int lane) {
    int n = lane & 15, half = lane >> 4;
    const u16* p = lds + (colBase + n) * stride + k0 + half * 8;
    union { uint4 q[2]; v16bf v; } f;
    f.q[0] = *reinterpret_cast<const uint4*>(p);
    f.q[1] = *reinterpret_cast<const uint4*>(p + 16);
    return f.v;
}

// ---------------------------------------------------------------- cast kernels

__global__ void cast_f32_to_bf16_kernel(const float* __restrict__ in,
                                        u16* __restrict__ out, size_t n) {
    size_t i = (size_t)blockIdx.x * blockDim.x + threadIdx.x;
    if (i < n) out[i] = f32_to_bf16(in[i]);
}

// out[n][k] = bf16(in[k][n]); in is [K][N] f32. 32x32 tiles, 256 threads.
__global__ __launch_bounds__(256)
void cast_transpose_kernel(const float* __restrict__ in, u16* __restrict__ out,
                           int K, int N) {
    __shared__ u16 t[32][33];
    const int n0 = blockIdx.x * 32, k0 = blockIdx.y * 32;
    const int tx = threadIdx.x & 31, ty = threadIdx.x >> 5;   // 32 x 8
#pragma unroll
    for (int i = 0; i < 32; i += 8)
        t[ty + i][tx] = f32_to_bf16(in[(size_t)(k0 + ty + i) * N + n0 + tx]);
    __syncthreads();
#pragma unroll
    for (int i = 0; i < 32; i += 8)
        out[(size_t)(n0 + ty + i) * K + k0 + tx] = t[tx][ty + i];
}

// ---------------------------------------------------------------- WMMA GEMM
// C[M,N] = A[M,K](bf16,rm) * Bt[N,K](bf16,rm = B^T) + bias.
//   EPI=0: out bf16 = acc + bias
//   EPI=1: out bf16 = gelu_exact(acc + bias)
//   EPI=2: out f32  = acc + bias + res
// Block 128x256, BK=32, 256 threads = 8 waves in 2(m) x 4(n); wave = 64x64
// (16 WMMA per K-step per wave). Double-buffered async LDS staging.

template <int EPI>
__global__ __launch_bounds__(256)
void gemm_bf16_wmma_kernel(const u16* __restrict__ A, const u16* __restrict__ Bt,
                           const float* __restrict__ bias,
                           const float* __restrict__ res,
                           void* __restrict__ out, int M, int N, int K) {
    __shared__ u16 As[2][128 * 40];   // [row M][K+pad]
    __shared__ u16 Bs[2][256 * 40];   // [row N][K+pad]

    const int tid  = threadIdx.x;
    const int lane = tid & 31;
    const int wave = tid >> 5;
    const int wm   = wave >> 2;        // 0..1 -> 64-row stripe
    const int wn   = wave & 3;         // 0..3 -> 64-col stripe
    const int n0   = blockIdx.x * 256;
    const int m0   = blockIdx.y * 128;

    v8f acc[4][4];
#pragma unroll
    for (int mt = 0; mt < 4; ++mt)
#pragma unroll
        for (int nt = 0; nt < 4; ++nt) acc[mt][nt] = v8f_zero();

    auto stage = [&](int buf, int k0) {
#pragma unroll
        for (int c = tid; c < 512; c += 256) {        // A: 128x32
            int row = c >> 2, kc = (c & 3) * 8;
            copy16(&As[buf][row * 40 + kc], A + (size_t)(m0 + row) * K + k0 + kc);
        }
#pragma unroll
        for (int c = tid; c < 1024; c += 256) {       // B^T: 256x32
            int row = c >> 2, kc = (c & 3) * 8;
            copy16(&Bs[buf][row * 40 + kc], Bt + (size_t)(n0 + row) * K + k0 + kc);
        }
    };

    stage(0, 0);
    stage_fence();

    const int nk = K / 32;
    for (int ki = 0; ki < nk; ++ki) {
        const int buf = ki & 1;
        if (ki + 1 < nk) stage(buf ^ 1, (ki + 1) * 32);   // overlap with WMMA

        v16bf bfr[4];
#pragma unroll
        for (int nt = 0; nt < 4; ++nt)
            bfr[nt] = load_bt_frag(&Bs[buf][0], wn * 64 + nt * 16, 40, 0, lane);
#pragma unroll
        for (int mt = 0; mt < 4; ++mt) {
            v16bf af = load_a_frag(&As[buf][0], wm * 64 + mt * 16, 40, 0, lane);
#pragma unroll
            for (int nt = 0; nt < 4; ++nt)
                acc[mt][nt] = wmma_bf16(af, bfr[nt], acc[mt][nt]);
        }
        stage_fence();
    }

    // epilogue: C layout -> lane(0-15)=col n, VGPR r -> row r + 8*half
    const int n = lane & 15, half = lane >> 4;
#pragma unroll
    for (int mt = 0; mt < 4; ++mt) {
#pragma unroll
        for (int nt = 0; nt < 4; ++nt) {
            int col = n0 + wn * 64 + nt * 16 + n;
            float bv = bias[col];
#pragma unroll
            for (int r = 0; r < 8; ++r) {
                int row = m0 + wm * 64 + mt * 16 + r + half * 8;
                float v = acc[mt][nt][r] + bv;
                size_t idx = (size_t)row * N + col;
                if (EPI == 2) {
                    ((float*)out)[idx] = v + res[idx];
                } else {
                    if (EPI == 1)
                        v = 0.5f * v * (1.0f + erff(v * 0.70710678118654752f));
                    ((u16*)out)[idx] = f32_to_bf16(v);
                }
            }
        }
    }
}

// ---------------------------------------------------------------- attention
// Flash-style, no causal mask (per reference). One block per (b,h,128 queries).

__global__ __launch_bounds__(256)
void attention_wmma_kernel(const u16* __restrict__ qkv, u16* __restrict__ out,
                           int Bb, int T, int C, int H) {
    (void)Bb; (void)H;
    __shared__ u16 Qs[128 * 72];   // [q][dh+pad]
    __shared__ u16 KVs[64 * 72];   // K phase: [key][dh]; V phase: [dh][key]
    __shared__ u16 Ps[128 * 72];   // [q][key+pad], bf16 probabilities

    const int tid  = threadIdx.x;
    const int lane = tid & 31;
    const int wave = tid >> 5;         // 8 waves, each owns 16 query rows
    const int b  = blockIdx.z;
    const int h  = blockIdx.y;
    const int t0 = blockIdx.x * 128;
    const int ld = 3 * C;

#pragma unroll
    for (int c = tid; c < 1024; c += 256) {
        int row = c >> 3, d8 = (c & 7) * 8;
        uint4 d = *reinterpret_cast<const uint4*>(
            qkv + (size_t)(b * T + t0 + row) * ld + h * 64 + d8);
        *reinterpret_cast<uint4*>(&Qs[row * 72 + d8]) = d;
    }
    __syncthreads();

    v16bf qa[2];
    qa[0] = load_a_frag(Qs, wave * 16, 72, 0, lane);
    qa[1] = load_a_frag(Qs, wave * 16, 72, 32, lane);

    v8f ot[4];
#pragma unroll
    for (int nt = 0; nt < 4; ++nt) ot[nt] = v8f_zero();
    float mrun[8], lrun[8];
#pragma unroll
    for (int r = 0; r < 8; ++r) { mrun[r] = -1e30f; lrun[r] = 0.f; }

    const float scale = 0.125f;        // 1/sqrt(64)
    const int n = lane & 15, half = lane >> 4;

    for (int j = 0; j < T / 64; ++j) {
        const int kt0 = j * 64;
        __syncthreads();
#pragma unroll
        for (int c = tid; c < 512; c += 256) {   // K tile: natural layout
            int row = c >> 3, d8 = (c & 7) * 8;
            uint4 d = *reinterpret_cast<const uint4*>(
                qkv + (size_t)(b * T + kt0 + row) * ld + C + h * 64 + d8);
            *reinterpret_cast<uint4*>(&KVs[row * 72 + d8]) = d;
        }
        __syncthreads();

        v8f st[4];
#pragma unroll
        for (int nt = 0; nt < 4; ++nt) st[nt] = v8f_zero();
#pragma unroll
        for (int s = 0; s < 2; ++s)
#pragma unroll
            for (int nt = 0; nt < 4; ++nt) {
                v16bf kb = load_bt_frag(KVs, nt * 16, 72, s * 32, lane);
                st[nt] = wmma_bf16(qa[s], kb, st[nt]);
            }

#pragma unroll
        for (int r = 0; r < 8; ++r) {
            float mx = fmaxf(fmaxf(st[0][r], st[1][r]), fmaxf(st[2][r], st[3][r]));
#pragma unroll
            for (int off = 1; off < 16; off <<= 1)
                mx = fmaxf(mx, __shfl_xor(mx, off, 32));
            mx *= scale;
            float mnew  = fmaxf(mrun[r], mx);
            float alpha = __expf(mrun[r] - mnew);
            float rsum = 0.f;
#pragma unroll
            for (int nt = 0; nt < 4; ++nt) {
                float p = __expf(st[nt][r] * scale - mnew);
                st[nt][r] = p;
                rsum += p;
            }
#pragma unroll
            for (int off = 1; off < 16; off <<= 1)
                rsum += __shfl_xor(rsum, off, 32);
#pragma unroll
            for (int nt = 0; nt < 4; ++nt) ot[nt][r] *= alpha;
            lrun[r] = lrun[r] * alpha + rsum;
            mrun[r] = mnew;
        }

#pragma unroll
        for (int nt = 0; nt < 4; ++nt)
#pragma unroll
            for (int r = 0; r < 8; ++r)
                Ps[(wave * 16 + r + half * 8) * 72 + nt * 16 + n] =
                    f32_to_bf16(st[nt][r]);

        __syncthreads();
#pragma unroll
        for (int c = tid; c < 512; c += 256) {   // V tile, transposed [dh][key]
            int key = c >> 3, d8 = (c & 7) * 8;
            uint4 d = *reinterpret_cast<const uint4*>(
                qkv + (size_t)(b * T + kt0 + key) * ld + 2 * C + h * 64 + d8);
            const u16* s = reinterpret_cast<const u16*>(&d);
#pragma unroll
            for (int i = 0; i < 8; ++i) KVs[(d8 + i) * 72 + key] = s[i];
        }
        __syncthreads();

#pragma unroll
        for (int s = 0; s < 2; ++s) {
            v16bf pa = load_a_frag(Ps, wave * 16, 72, s * 32, lane);
#pragma unroll
            for (int nt = 0; nt < 4; ++nt) {
                v16bf vb = load_bt_frag(KVs, nt * 16, 72, s * 32, lane);
                ot[nt] = wmma_bf16(pa, vb, ot[nt]);
            }
        }
    }

#pragma unroll
    for (int nt = 0; nt < 4; ++nt)
#pragma unroll
        for (int r = 0; r < 8; ++r) {
            int row = t0 + wave * 16 + r + half * 8;
            float o = ot[nt][r] / lrun[r];
            out[(size_t)(b * T + row) * C + h * 64 + nt * 16 + n] = f32_to_bf16(o);
        }
}

// ---------------------------------------------------------------- layernorm

__global__ __launch_bounds__(256)
void layernorm_kernel(const float* __restrict__ x, const float* __restrict__ w,
                      const float* __restrict__ bias, u16* __restrict__ out, int C) {
    __shared__ float ssum[8], ssum2[8];
    const int row = blockIdx.x;
    const float* xr = x + (size_t)row * C;
    float4 v4 = reinterpret_cast<const float4*>(xr)[threadIdx.x];
    float v[4] = {v4.x, v4.y, v4.z, v4.w};
    float s = 0.f, s2 = 0.f;
#pragma unroll
    for (int i = 0; i < 4; ++i) { s += v[i]; s2 += v[i] * v[i]; }
#pragma unroll
    for (int off = 1; off < 32; off <<= 1) {
        s  += __shfl_xor(s, off, 32);
        s2 += __shfl_xor(s2, off, 32);
    }
    if ((threadIdx.x & 31) == 0) {
        ssum[threadIdx.x >> 5]  = s;
        ssum2[threadIdx.x >> 5] = s2;
    }
    __syncthreads();
    float tot = 0.f, tot2 = 0.f;
#pragma unroll
    for (int i = 0; i < 8; ++i) { tot += ssum[i]; tot2 += ssum2[i]; }
    const float inv = 1.0f / (float)C;
    float mean = tot * inv;
    float var  = tot2 * inv - mean * mean;
    float rs   = rsqrtf(var + 1e-5f);
    int base = threadIdx.x * 4;
#pragma unroll
    for (int i = 0; i < 4; ++i)
        out[(size_t)row * C + base + i] =
            f32_to_bf16((v[i] - mean) * rs * w[base + i] + bias[base + i]);
}

// ---------------------------------------------------------------- launcher

extern "C" void kernel_launch(void* const* d_in, const int* in_sizes, int n_in,
                              void* d_out, int out_size, void* d_ws, size_t ws_size,
                              hipStream_t stream) {
    (void)in_sizes; (void)n_in; (void)out_size; (void)ws_size;
    const float* x      = (const float*)d_in[0];
    const float* w_attn = (const float*)d_in[1];
    const float* b_attn = (const float*)d_in[2];
    const float* w_o    = (const float*)d_in[3];
    const float* b_o    = (const float*)d_in[4];
    const float* ln_w   = (const float*)d_in[5];
    const float* ln_b   = (const float*)d_in[6];
    const float* w_fc1  = (const float*)d_in[7];
    const float* b_fc1  = (const float*)d_in[8];
    const float* w_fc2  = (const float*)d_in[9];
    const float* b_fc2  = (const float*)d_in[10];

    const int B = 4, T = 2048, C = 1024;
    const int M = B * T;                      // 8192

    size_t off = 0;
    char* ws = (char*)d_ws;
    auto carve = [&](size_t bytes) -> void* {
        void* p = ws + off;
        off += (bytes + 255) & ~(size_t)255;
        return p;
    };
    u16*   xb      = (u16*)  carve((size_t)M * C * 2);
    u16*   wattnT  = (u16*)  carve((size_t)C * 3 * C * 2);   // [3C][C]
    u16*   woT     = (u16*)  carve((size_t)C * C * 2);       // [C][C]
    u16*   wfc1T   = (u16*)  carve((size_t)C * 4 * C * 2);   // [4C][C]
    u16*   wfc2T   = (u16*)  carve((size_t)4 * C * C * 2);   // [C][4C]
    u16*   qkvb    = (u16*)  carve((size_t)M * 3 * C * 2);
    u16*   attnb   = (u16*)  carve((size_t)M * C * 2);
    float* x1      = (float*)carve((size_t)M * C * 4);
    u16*   lnb     = (u16*)  carve((size_t)M * C * 2);
    u16*   h1b     = (u16*)  carve((size_t)M * 4 * C * 2);

    cast_f32_to_bf16_kernel<<<(unsigned)(((size_t)M * C + 511) / 512), 512, 0, stream>>>(
        x, xb, (size_t)M * C);
    {   // weight casts with transpose: out[N][K] from in[K][N]
        dim3 g1(3 * C / 32, C / 32);
        cast_transpose_kernel<<<g1, 256, 0, stream>>>(w_attn, wattnT, C, 3 * C);
        dim3 g2(C / 32, C / 32);
        cast_transpose_kernel<<<g2, 256, 0, stream>>>(w_o, woT, C, C);
        dim3 g3(4 * C / 32, C / 32);
        cast_transpose_kernel<<<g3, 256, 0, stream>>>(w_fc1, wfc1T, C, 4 * C);
        dim3 g4(C / 32, 4 * C / 32);
        cast_transpose_kernel<<<g4, 256, 0, stream>>>(w_fc2, wfc2T, 4 * C, C);
    }

    // 1) qkv = x @ w_attn + b_attn            [M, 3C] bf16
    {
        dim3 g(3 * C / 256, M / 128);
        gemm_bf16_wmma_kernel<0><<<g, 256, 0, stream>>>(
            xb, wattnT, b_attn, nullptr, qkvb, M, 3 * C, C);
    }
    // 2) attention                             [M, C] bf16
    {
        dim3 g(T / 128, 16, B);
        attention_wmma_kernel<<<g, 256, 0, stream>>>(qkvb, attnb, B, T, C, 16);
    }
    // 3) x1 = x + attn @ w_o + b_o             [M, C] f32
    {
        dim3 g(C / 256, M / 128);
        gemm_bf16_wmma_kernel<2><<<g, 256, 0, stream>>>(
            attnb, woT, b_o, x, x1, M, C, C);
    }
    // 4) ln = layer_norm(x1)                   [M, C] bf16
    layernorm_kernel<<<M, 256, 0, stream>>>(x1, ln_w, ln_b, lnb, C);
    // 5) h1 = gelu(ln @ w_fc1 + b_fc1)         [M, 4C] bf16
    {
        dim3 g(4 * C / 256, M / 128);
        gemm_bf16_wmma_kernel<1><<<g, 256, 0, stream>>>(
            lnb, wfc1T, b_fc1, nullptr, h1b, M, 4 * C, C);
    }
    // 6) out = x1 + h1 @ w_fc2 + b_fc2         [M, C] f32
    {
        dim3 g(C / 256, M / 128);
        gemm_bf16_wmma_kernel<2><<<g, 256, 0, stream>>>(
            h1b, wfc2T, b_fc2, x1, d_out, M, C, 4 * C);
    }
}
